// MultiHeadSelfAttention_85839216378297
// MI455X (gfx1250) — matrix operand order repaired and checked by reading, not compiled, not run on hardware
//
#include <hip/hip_runtime.h>
#include <stdint.h>

// ---------------------------------------------------------------------------
// MultiHeadSelfAttention for MI455X (gfx1250).
// bf16 WMMA GEMMs + flash attention, TDM (tensor_load_to_lds) tile DMA,
// double-buffered LDS with one barrier per k-step.
// B=2, C=1024, W=2048, H=16, D=64.
// ---------------------------------------------------------------------------

typedef __attribute__((ext_vector_type(16))) __bf16        v16bf;
typedef __attribute__((ext_vector_type(8)))  float         v8f;
typedef __attribute__((ext_vector_type(8)))  unsigned short u16x8;
typedef __attribute__((ext_vector_type(4)))  unsigned short u16x4;
typedef __attribute__((ext_vector_type(4)))  unsigned int   v4u;
typedef __attribute__((ext_vector_type(8)))  int            v8i;
typedef __attribute__((ext_vector_type(4)))  int            v4i;

union ABFrag  { u16x8 h[2]; v16bf v; };
union ABFragS { unsigned short s[16]; v16bf v; };

__device__ __forceinline__ unsigned short f32_to_bf16(float f) {
  unsigned int u = __float_as_uint(f);
  u += 0x7fffu + ((u >> 16) & 1u);   // round-to-nearest-even
  return (unsigned short)(u >> 16);
}

// ---------------------------------------------------------------------------
// TDM: DMA a 2D bf16 tile (tileW x tileH elements, row stride = strideElems)
// from global memory into LDS at ldsAddr, optionally inserting 1 DWORD of LDS
// padding after every 2^(padInterval+1) DWORDs (i.e. per row) to de-conflict
// banks. D# encoding per CDNA5 ISA.
// This toolchain exposes the 6-arg builtin: (v4u, v8i, v4i, v4i, v8i, cpol).
// Issue from ONE wave only; completion tracked with TENSORcnt.
// ---------------------------------------------------------------------------
__device__ __forceinline__ void tdm_load_2d_bf16(
    unsigned ldsAddr, const void* gptr,
    unsigned tensorW, unsigned tensorH,
    unsigned tileW, unsigned tileH,
    unsigned long long strideElems, int padInterval) {
  unsigned long long ga = (unsigned long long)gptr;
  v4u g0;
  g0[0] = 1u;                                        // count=1 (valid user D#)
  g0[1] = ldsAddr;                                   // lds_addr (bytes)
  g0[2] = (unsigned)ga;                              // global_addr[31:0]
  g0[3] = (unsigned)(ga >> 32) | (2u << 30);         // global_addr[56:32] | type=2
  int w0 = 0x00010000;                               // data_size=1 -> 2 bytes
  if (padInterval >= 0) w0 |= (1 << 20) | (padInterval << 22);  // pad_amount=0 (1 DW)
  v8i g1;
  g1[0] = w0;
  g1[1] = (int)((tensorW & 0xFFFFu) << 16);                          // dim0[15:0]
  g1[2] = (int)(((tensorW >> 16) & 0xFFFFu) | ((tensorH & 0xFFFFu) << 16));
  g1[3] = (int)(((tensorH >> 16) & 0xFFFFu) | (tileW << 16));        // tile_dim0
  g1[4] = (int)tileH;                                                // tile_dim1
  g1[5] = (int)(strideElems & 0xFFFFFFFFull);                        // dim0_stride lo
  g1[6] = (int)((strideElems >> 32) & 0xFFFFull);                    // dim0_stride hi
  g1[7] = 0;
  v4i g2; g2[0] = 1; g2[1] = 0; g2[2] = 0; g2[3] = 0;                // dim2=1 (unused)
  v4i g3; g3[0] = 0; g3[1] = 0; g3[2] = 0; g3[3] = 0;
  v8i g4;                                                            // aux group (zero)
#pragma unroll
  for (int i = 0; i < 8; ++i) g4[i] = 0;
  __builtin_amdgcn_tensor_load_to_lds(g0, g1, g2, g3, g4, 0);
}

// B-fragment (32x16, 16-bit) gather from a plain row-major LDS tile:
// lane = ((k/8)&1)*16 + n, element e <-> k = (lane/16)*8 + e + (e&8).
__device__ __forceinline__ v16bf load_bfrag_strided(
    const unsigned short* base, int rowStride, int lane) {
  ABFragS f;
  const unsigned short* p = base + (lane >> 4) * 8 * rowStride + (lane & 15);
#pragma unroll
  for (int e = 0; e < 16; ++e) f.s[e] = p[(e + (e & 8)) * rowStride];
  return f.v;
}

// ---------------------------------------------------------------------------
// f32 -> bf16 conversion (vectorized 4/thread)
// ---------------------------------------------------------------------------
__global__ __launch_bounds__(256) void cvt_f32_bf16_kernel(
    const float* __restrict__ src, unsigned short* __restrict__ dst, int n) {
  int i = (blockIdx.x * 256 + threadIdx.x) * 4;
  if (i < n) {
    float4 f = *(const float4*)(src + i);
    u16x4 r;
    r[0] = f32_to_bf16(f.x);
    r[1] = f32_to_bf16(f.y);
    r[2] = f32_to_bf16(f.z);
    r[3] = f32_to_bf16(f.w);
    *(u16x4*)(dst + i) = r;
  }
}

// ---------------------------------------------------------------------------
// bf16 WMMA GEMM:  Out[b] = A(MxK) * Bmat[b](KxN) + bias
// Tile 128x128x32, 256 thr = 8 waves (2x4), wave computes 64x32.
// A tile: manual fragment-flat LDS (vector stores, ds_load_b128 frag loads).
// B tile: TDM DMA into plain padded row-major LDS (stride 130 shorts),
//         B-fragments gathered with 16 ds_load_u16 (bank-conflict free).
// Double buffered, single barrier per k-step; TDM issued 1 step ahead.
// ---------------------------------------------------------------------------
#define GEMM_BM 128
#define GEMM_BN 128
#define GEMM_BK 32
#define BS_STRIDE 130   // 128 shorts + 1 DWORD TDM pad

__global__ __launch_bounds__(256) void gemm_bf16_wmma_kernel(
    const unsigned short* __restrict__ A,
    const unsigned short* __restrict__ Bmat,
    const float* __restrict__ bias,
    unsigned short* __restrict__ outB,
    float* __restrict__ outF,
    const float* __restrict__ resid,
    const float* __restrict__ gammaPtr,
    int M, int N, int K, long long batchStride) {
  __shared__ __align__(16) unsigned short As[2][8 * 512];
  __shared__ __align__(16) unsigned short Bs[2][GEMM_BK * BS_STRIDE];

  const int tid   = threadIdx.x;
  const int lane  = tid & 31;
  const int wid   = tid >> 5;
  const int waveM = wid >> 2;           // 0..1
  const int waveN = wid & 3;            // 0..3
  const int mBase = blockIdx.y * GEMM_BM;
  const int nBase = blockIdx.x * GEMM_BN;
  const long long boff = (long long)blockIdx.z * batchStride;
  Bmat += boff;

  // per-slot A staging geometry (loop invariant)
  int aRow[2], aK0[2], aLds[2];
#pragma unroll
  for (int i = 0; i < 2; ++i) {
    int slot = tid + 256 * i;           // 512 slots = 128 rows x 4 k-groups
    aRow[i] = slot >> 2;
    aK0[i]  = (slot & 3) << 3;
    int mt = aRow[i] >> 4, m = aRow[i] & 15;
    int half = (aK0[i] >> 3) & 1;
    int e0   = (aK0[i] >> 4) << 3;
    aLds[i]  = mt * 512 + (half * 16 + m) * 16 + e0;
  }

  v8f acc[4][2] = {};
  const int nTiles = K / GEMM_BK;

  // ---- prologue: tile 0 ---------------------------------------------------
#pragma unroll
  for (int i = 0; i < 2; ++i) {
    u16x8 d = *(const u16x8*)(A + (size_t)(mBase + aRow[i]) * K + aK0[i]);
    *(u16x8*)(&As[0][aLds[i]]) = d;
  }
  if (wid == 0) {
    tdm_load_2d_bf16((unsigned)(uintptr_t)&Bs[0][0], Bmat + nBase,
                     (unsigned)N, (unsigned)K, GEMM_BN, GEMM_BK,
                     (unsigned long long)N, /*padInterval=*/5);
    __builtin_amdgcn_s_wait_tensorcnt(0);
  }

  for (int t = 0; t < nTiles; ++t) {
    const int cur = t & 1, nxt = cur ^ 1;
    __syncthreads();                    // publish tile t to everyone

    const bool hasNext = (t + 1) < nTiles;
    const int ktN = (t + 1) * GEMM_BK;
    u16x8 aR[2];
    if (hasNext) {
      if (wid == 0)
        tdm_load_2d_bf16((unsigned)(uintptr_t)&Bs[nxt][0],
                         Bmat + (size_t)ktN * N + nBase,
                         (unsigned)N, (unsigned)(K - ktN), GEMM_BN, GEMM_BK,
                         (unsigned long long)N, 5);
#pragma unroll
      for (int i = 0; i < 2; ++i)
        aR[i] = *(const u16x8*)(A + (size_t)(mBase + aRow[i]) * K + ktN + aK0[i]);
    }

    // ---- compute on tile t -----------------------------------------------
    ABFrag aF[4];
#pragma unroll
    for (int mt = 0; mt < 4; ++mt) {
      const u16x8* p = (const u16x8*)(&As[cur][(waveM * 4 + mt) * 512 + lane * 16]);
      aF[mt].h[0] = p[0]; aF[mt].h[1] = p[1];
    }
#pragma unroll
    for (int nt = 0; nt < 2; ++nt) {
      v16bf bf = load_bfrag_strided(&Bs[cur][waveN * 32 + nt * 16], BS_STRIDE, lane);
#pragma unroll
      for (int mt = 0; mt < 4; ++mt)
        acc[mt][nt] = __builtin_amdgcn_wmma_f32_16x16x32_bf16(
            false, aF[mt].v, false, bf, (short)0, acc[mt][nt], false, false);
    }

    if (hasNext) {
#pragma unroll
      for (int i = 0; i < 2; ++i) *(u16x8*)(&As[nxt][aLds[i]]) = aR[i];
      if (wid == 0) __builtin_amdgcn_s_wait_tensorcnt(0);
    }
  }

  // ---- epilogue ----------------------------------------------------------
  const int halfL = lane >> 4;
  const int colL  = lane & 15;
  const float g = outF ? gammaPtr[0] : 0.0f;
#pragma unroll
  for (int mt = 0; mt < 4; ++mt) {
#pragma unroll
    for (int nt = 0; nt < 2; ++nt) {
#pragma unroll
      for (int vv = 0; vv < 8; ++vv) {
        int m = mBase + waveM * 64 + mt * 16 + halfL * 8 + vv;
        int n = nBase + waveN * 32 + nt * 16 + colL;
        float val = acc[mt][nt][vv] + bias[m];
        size_t idx = (size_t)m * N + n;
        if (outF) outF[boff + idx] = g * val + resid[boff + idx];
        else      outB[boff + idx] = f32_to_bf16(val);
      }
    }
  }
}

// ---------------------------------------------------------------------------
// Flash attention: WG = 4 waves / 64 queries of one (b,h).
// K tiles (64d x 64key) DMA'd by TDM into plain padded LDS (stride 66),
// V tiles manually staged into fragment-flat LDS (vectorized both ways).
// Double-buffered, one barrier per 64-key block.
// ---------------------------------------------------------------------------
#define KS_STRIDE 66    // 64 shorts + 1 DWORD TDM pad

__global__ __launch_bounds__(128) void flash_attn_bf16_kernel(
    const unsigned short* __restrict__ qg,
    const unsigned short* __restrict__ kg,
    const unsigned short* __restrict__ vg,
    unsigned short* __restrict__ og) {
  constexpr int W = 2048, D = 64, H = 16;
  constexpr float SCALE = 0.125f;  // 1/sqrt(64)

  __shared__ __align__(16) unsigned short Qs[8 * 512];           // A-layout
  __shared__ __align__(16) unsigned short Ks[2][64 * KS_STRIDE]; // plain [d][key]
  __shared__ __align__(16) unsigned short Vs[2][8 * 512];        // B-frag layout
  __shared__ __align__(16) unsigned short Ps[4 * 1024];          // per-wave A-layout

  const int tid  = threadIdx.x;
  const int lane = tid & 31;
  const int wv   = tid >> 5;
  const int bh   = blockIdx.z * H + blockIdx.y;
  const int qBase = blockIdx.x * 64;
  const size_t base = (size_t)bh * D * W;
  const int halfL = lane >> 4;
  const int colL  = lane & 15;

  // per-slot V staging geometry (loop invariant): slot = d*8 + keygroup
  int vD[4], vK0[4], vLds[4];
#pragma unroll
  for (int i = 0; i < 4; ++i) {
    int slot = tid + 128 * i;           // 512 slots
    vD[i]  = slot >> 3;
    vK0[i] = (slot & 7) << 3;
    int kl = vK0[i] & 31;
    vLds[i] = ((vK0[i] >> 5) * 4 + (vD[i] >> 4)) * 512 +
              (((kl >> 3) & 1) * 16 + (vD[i] & 15)) * 16 + ((kl >> 4) << 3);
  }

  // ---- stage Q once (A-fragment layout) ----------------------------------
#pragma unroll
  for (int i = 0; i < 4; ++i) {
    int slot = tid + 128 * i;
    int d = slot >> 3, q0 = (slot & 7) << 3;
    u16x8 val = *(const u16x8*)(qg + base + (size_t)d * W + qBase + q0);
    int ks = d >> 5, dl = d & 31;
    int ha = (dl & 15) >> 3;
    int e  = ((dl >> 4) << 3) + (dl & 7);
#pragma unroll
    for (int j = 0; j < 8; ++j) {
      int qq = q0 + j;
      Qs[((qq >> 4) * 2 + ks) * 512 + (ha * 16 + (qq & 15)) * 16 + e] = val[j];
    }
  }

  // ---- prologue: key-block 0 ---------------------------------------------
#pragma unroll
  for (int i = 0; i < 4; ++i) {
    u16x8 val = *(const u16x8*)(vg + base + (size_t)vD[i] * W + vK0[i]);
    *(u16x8*)(&Vs[0][vLds[i]]) = val;
  }
  if (wv == 0) {
    tdm_load_2d_bf16((unsigned)(uintptr_t)&Ks[0][0], kg + base,
                     (unsigned)W, (unsigned)D, 64, 64,
                     (unsigned long long)W, /*padInterval=*/4);
    __builtin_amdgcn_s_wait_tensorcnt(0);
  }
  __syncthreads();

  ABFrag qF[2];
#pragma unroll
  for (int ks = 0; ks < 2; ++ks) {
    const u16x8* p = (const u16x8*)(&Qs[(wv * 2 + ks) * 512 + lane * 16]);
    qF[ks].h[0] = p[0]; qF[ks].h[1] = p[1];
  }

  v8f O[4] = {};
  float mrow[8], lrow[8];
#pragma unroll
  for (int vv = 0; vv < 8; ++vv) { mrow[vv] = -1e30f; lrow[vv] = 0.0f; }

  constexpr int NKB = W / 64;
  for (int t = 0; t < NKB; ++t) {
    const int cur = t & 1, nxt = cur ^ 1;
    if (t) __syncthreads();             // publish tile t (t=0 published above)

    const bool hasNext = (t + 1) < NKB;
    const int kbN = (t + 1) * 64;
    u16x8 vR[4];
    if (hasNext) {
      if (wv == 0)
        tdm_load_2d_bf16((unsigned)(uintptr_t)&Ks[nxt][0], kg + base + kbN,
                         (unsigned)(W - kbN), (unsigned)D, 64, 64,
                         (unsigned long long)W, 4);
#pragma unroll
      for (int i = 0; i < 4; ++i)
        vR[i] = *(const u16x8*)(vg + base + (size_t)vD[i] * W + kbN + vK0[i]);
    }

    // ---- S = Q * K^T : gather K B-frags from plain [d][key] tile ---------
    v8f S[4];
#pragma unroll
    for (int tt = 0; tt < 4; ++tt) {
      v8f s = {};
#pragma unroll
      for (int ks = 0; ks < 2; ++ks) {
        v16bf kf = load_bfrag_strided(&Ks[cur][(ks * 32) * KS_STRIDE + tt * 16],
                                      KS_STRIDE, lane);
        s = __builtin_amdgcn_wmma_f32_16x16x32_bf16(
            false, qF[ks].v, false, kf, (short)0, s, false, false);
      }
      S[tt] = s;
    }

    // ---- streaming softmax ----------------------------------------------
    float alpha[8];
#pragma unroll
    for (int vv = 0; vv < 8; ++vv) {
      float mx = -1e30f;
#pragma unroll
      for (int tt = 0; tt < 4; ++tt) { S[tt][vv] *= SCALE; mx = fmaxf(mx, S[tt][vv]); }
#pragma unroll
      for (int off = 1; off < 16; off <<= 1)
        mx = fmaxf(mx, __shfl_xor(mx, off, 32));
      float nm = fmaxf(mrow[vv], mx);
      alpha[vv] = __expf(mrow[vv] - nm);
      mrow[vv] = nm;
      float ls = 0.0f;
#pragma unroll
      for (int tt = 0; tt < 4; ++tt) {
        float pexp = __expf(S[tt][vv] - nm);
        S[tt][vv] = pexp;
        ls += pexp;
      }
#pragma unroll
      for (int off = 1; off < 16; off <<= 1)
        ls += __shfl_xor(ls, off, 32);
      lrow[vv] = lrow[vv] * alpha[vv] + ls;
    }

    // ---- P -> per-wave LDS (A-layout), reload as fragments ---------------
#pragma unroll
    for (int tt = 0; tt < 4; ++tt) {
#pragma unroll
      for (int vv = 0; vv < 8; ++vv) {
        int e  = (tt & 1) * 8 + (colL & 7);
        int lp = (colL >> 3) * 16 + halfL * 8 + vv;
        Ps[wv * 1024 + (tt >> 1) * 512 + lp * 16 + e] = f32_to_bf16(S[tt][vv]);
      }
    }
    ABFrag pF[2];
#pragma unroll
    for (int ks = 0; ks < 2; ++ks) {
      const u16x8* p = (const u16x8*)(&Ps[wv * 1024 + ks * 512 + lane * 16]);
      pF[ks].h[0] = p[0]; pF[ks].h[1] = p[1];
    }

    // ---- O = diag(alpha)*O + P*V ----------------------------------------
#pragma unroll
    for (int n = 0; n < 4; ++n)
#pragma unroll
      for (int vv = 0; vv < 8; ++vv) O[n][vv] *= alpha[vv];
#pragma unroll
    for (int n = 0; n < 4; ++n) {
#pragma unroll
      for (int ks = 0; ks < 2; ++ks) {
        ABFrag vf;
        const u16x8* p = (const u16x8*)(&Vs[cur][(ks * 4 + n) * 512 + lane * 16]);
        vf.h[0] = p[0]; vf.h[1] = p[1];
        O[n] = __builtin_amdgcn_wmma_f32_16x16x32_bf16(
            false, pF[ks].v, false, vf.v, (short)0, O[n], false, false);
      }
    }

    if (hasNext) {
#pragma unroll
      for (int i = 0; i < 4; ++i) *(u16x8*)(&Vs[nxt][vLds[i]]) = vR[i];
      if (wv == 0) __builtin_amdgcn_s_wait_tensorcnt(0);
    }
  }

  // ---- finalize & store ---------------------------------------------------
#pragma unroll
  for (int n = 0; n < 4; ++n) {
#pragma unroll
    for (int vv = 0; vv < 8; ++vv) {
      float val = O[n][vv] / lrow[vv];
      int d    = n * 16 + colL;
      int qAbs = qBase + wv * 16 + halfL * 8 + vv;
      og[base + (size_t)d * W + qAbs] = f32_to_bf16(val);
    }
  }
}

// ---------------------------------------------------------------------------
// Host launcher
// ---------------------------------------------------------------------------
extern "C" void kernel_launch(void* const* d_in, const int* in_sizes, int n_in,
                              void* d_out, int out_size, void* d_ws, size_t ws_size,
                              hipStream_t stream) {
  constexpr int B = 2, C = 1024, W = 2048, H = 16;
  const float* x     = (const float*)d_in[0];
  const float* Wq    = (const float*)d_in[1];
  const float* bq    = (const float*)d_in[2];
  const float* Wk    = (const float*)d_in[3];
  const float* bk    = (const float*)d_in[4];
  const float* Wv    = (const float*)d_in[5];
  const float* bv    = (const float*)d_in[6];
  const float* Wfc   = (const float*)d_in[7];
  const float* bfc   = (const float*)d_in[8];
  const float* gamma = (const float*)d_in[9];

  char* ws = (char*)d_ws;
  size_t off = 0;
  auto take = [&](size_t bytes) { void* p = ws + off; off += bytes; return p; };
  const size_t xBytes = (size_t)B * C * W * 2;
  const size_t wBytes = (size_t)C * C * 2;
  unsigned short* xb   = (unsigned short*)take(xBytes);
  unsigned short* wqb  = (unsigned short*)take(wBytes);
  unsigned short* wkb  = (unsigned short*)take(wBytes);
  unsigned short* wvb  = (unsigned short*)take(wBytes);
  unsigned short* wfcb = (unsigned short*)take(wBytes);
  unsigned short* qb   = (unsigned short*)take(xBytes);
  unsigned short* kb   = (unsigned short*)take(xBytes);
  unsigned short* vb   = (unsigned short*)take(xBytes);
  unsigned short* attb = (unsigned short*)take(xBytes);

  // 1) f32 -> bf16 conversions
  int nX = B * C * W;
  cvt_f32_bf16_kernel<<<nX / 1024, 256, 0, stream>>>(x, xb, nX);
  int nW = C * C;
  cvt_f32_bf16_kernel<<<nW / 1024, 256, 0, stream>>>(Wq,  wqb,  nW);
  cvt_f32_bf16_kernel<<<nW / 1024, 256, 0, stream>>>(Wk,  wkb,  nW);
  cvt_f32_bf16_kernel<<<nW / 1024, 256, 0, stream>>>(Wv,  wvb,  nW);
  cvt_f32_bf16_kernel<<<nW / 1024, 256, 0, stream>>>(Wfc, wfcb, nW);

  // 2) QKV projections (bf16 out, bias fused)
  dim3 gGemm(W / GEMM_BN, C / GEMM_BM, B);
  long long bStride = (long long)C * W;
  gemm_bf16_wmma_kernel<<<gGemm, 256, 0, stream>>>(
      wqb, xb, bq, qb, nullptr, nullptr, nullptr, C, W, C, bStride);
  gemm_bf16_wmma_kernel<<<gGemm, 256, 0, stream>>>(
      wkb, xb, bk, kb, nullptr, nullptr, nullptr, C, W, C, bStride);
  gemm_bf16_wmma_kernel<<<gGemm, 256, 0, stream>>>(
      wvb, xb, bv, vb, nullptr, nullptr, nullptr, C, W, C, bStride);

  // 3) flash attention
  flash_attn_bf16_kernel<<<dim3(W / 64, H, B), 128, 0, stream>>>(qb, kb, vb, attb);

  // 4) FC projection, fused epilogue: out = gamma*(Wfc*att + bfc) + x
  gemm_bf16_wmma_kernel<<<gGemm, 256, 0, stream>>>(
      wfcb, attb, bfc, nullptr, (float*)d_out, x, gamma, C, W, C, bStride);
}
